// SparseAttention_20229295964911
// MI455X (gfx1250) — compile-verified, matrix operand order checked by reading, not run on hardware
//
#include <hip/hip_runtime.h>

// ---------------------------------------------------------------------------
// SparseAttention for MI455X (gfx1250, wave32, WMMA)
// S=512, B=16, H=1024, N=16 heads, HD=64
// All GEMMs on v_wmma_f32_16x16x32_bf16 (bf16 in, fp32 accumulate).
// K-loops are 2x-unrolled ping-pong pipelines (no register-rotation copies);
// V is async-staged global->LDS (ASYNCcnt) overlapping scores + softmax.
// ---------------------------------------------------------------------------

#define S_  512
#define B_  16
#define H_  1024
#define N_  16
#define HD_ 64

typedef __bf16 bf16_t;
typedef __attribute__((ext_vector_type(16))) __bf16 v16bf;
typedef __attribute__((ext_vector_type(8)))  __bf16 v8bf;
typedef __attribute__((ext_vector_type(8)))  float  v8f;

#define WMMA_BF16(a, b, c) \
    __builtin_amdgcn_wmma_f32_16x16x32_bf16(false, (a), false, (b), (short)0, (c), false, false)

// Load a 16x32 bf16 WMMA operand fragment from a row-major matrix.
// A operand: row0 = tile row. B operand: pass the TRANSPOSED matrix (rows = B cols).
// ISA 7.12.2 (16-bit A 16x32): lanes 0-15 hold K {k0+0..7, k0+16..23},
// lanes 16-31 hold K {k0+8..15, k0+24..31}, row = lane & 15.
__device__ __forceinline__ v16bf load_frag(const bf16_t* __restrict__ mat,
                                           int ld, int row0, int k0) {
    int lane = threadIdx.x & 31;
    int r  = row0 + (lane & 15);
    int kk = k0 + ((lane >> 4) << 3);
    const bf16_t* p = mat + (size_t)r * ld + kk;
    v8bf lo = *reinterpret_cast<const v8bf*>(p);
    v8bf hi = *reinterpret_cast<const v8bf*>(p + 16);
    v16bf o;
#pragma unroll
    for (int i = 0; i < 8; ++i) { o[i] = lo[i]; o[i + 8] = hi[i]; }
    return o;
}

// ---------------- Kernel 1a: hidden (S,B,H) f32 -> x (B,S,H) bf16 ----------
__global__ void conv_x_kernel(const float* __restrict__ hs, bf16_t* __restrict__ xT) {
    size_t i = (size_t)blockIdx.x * 256 + threadIdx.x;
    if (i >= (size_t)S_ * B_ * H_) return;
    int h = (int)(i % H_);
    size_t sb = i / H_;
    int b = (int)(sb % B_);
    int s = (int)(sb / B_);
    xT[((size_t)b * S_ + s) * H_ + h] = (bf16_t)hs[i];
}

// ---------------- Kernel 1b: W (rows x cols) f32 -> W^T (cols x rows) bf16 -
__global__ void conv_wT_kernel(const float* __restrict__ W, bf16_t* __restrict__ WT,
                               int rows, int cols) {
    size_t i = (size_t)blockIdx.x * 256 + threadIdx.x;
    if (i >= (size_t)rows * cols) return;
    int h = (int)(i % rows);
    size_t c = i / rows;
    WT[c * rows + h] = (bf16_t)W[(size_t)h * cols + c];
}

// ---------------- Kernel 2: QKV GEMM (ping-pong pipelined) -----------------
__global__ void __launch_bounds__(32)
qkv_gemm_kernel(const bf16_t* __restrict__ xT, const bf16_t* __restrict__ WqkvT,
                const float* __restrict__ bqkv,
                bf16_t* __restrict__ qb, bf16_t* __restrict__ kb, bf16_t* __restrict__ vT) {
    int strip = blockIdx.x;     // 0..47  (64-wide output strip of 3H)
    int st    = blockIdx.y;     // 0..31  (16-row tile of S)
    int b     = blockIdx.z;     // 0..15
    int lane  = threadIdx.x;
    const bf16_t* xb = xT + (size_t)b * S_ * H_;
    int s0 = st * 16;
    int c0 = strip * 64;

    v8f acc[4] = {};
    v16bf a0 = load_frag(xb, H_, s0, 0);
    v16bf b0[4];
#pragma unroll
    for (int t = 0; t < 4; ++t) b0[t] = load_frag(WqkvT, H_, c0 + 16 * t, 0);
    v16bf a1 = load_frag(xb, H_, s0, 32);
    v16bf b1[4];
#pragma unroll
    for (int t = 0; t < 4; ++t) b1[t] = load_frag(WqkvT, H_, c0 + 16 * t, 32);

    for (int k0 = 0; k0 < H_ - 64; k0 += 64) {
#pragma unroll
        for (int t = 0; t < 4; ++t) acc[t] = WMMA_BF16(a0, b0[t], acc[t]);
        a0 = load_frag(xb, H_, s0, k0 + 64);            // refill after last use
#pragma unroll
        for (int t = 0; t < 4; ++t) b0[t] = load_frag(WqkvT, H_, c0 + 16 * t, k0 + 64);
#pragma unroll
        for (int t = 0; t < 4; ++t) acc[t] = WMMA_BF16(a1, b1[t], acc[t]);
        a1 = load_frag(xb, H_, s0, k0 + 96);
#pragma unroll
        for (int t = 0; t < 4; ++t) b1[t] = load_frag(WqkvT, H_, c0 + 16 * t, k0 + 96);
    }
#pragma unroll
    for (int t = 0; t < 4; ++t) acc[t] = WMMA_BF16(a0, b0[t], acc[t]);
#pragma unroll
    for (int t = 0; t < 4; ++t) acc[t] = WMMA_BF16(a1, b1[t], acc[t]);

    int colL = lane & 15;
    int rowAdd = (lane >> 4) << 3;
#pragma unroll
    for (int t = 0; t < 4; ++t) {
#pragma unroll
        for (int r = 0; r < 8; ++r) {
            int sg = s0 + r + rowAdd;
            int c  = c0 + 16 * t + colL;
            float v = acc[t][r] + bqkv[c];
            bf16_t bv = (bf16_t)v;
            int n  = c / (3 * HD_);
            int r2 = c % (3 * HD_);
            size_t hb = (size_t)b * N_ + n;
            if (r2 < HD_)            qb[(hb * S_ + sg) * HD_ + r2]             = bv;
            else if (r2 < 2 * HD_)   kb[(hb * S_ + sg) * HD_ + (r2 - HD_)]     = bv;
            else                     vT[(hb * HD_ + (r2 - 2 * HD_)) * S_ + sg] = bv;
        }
    }
}

// ---------------- Kernel 3: sparse attention -------------------------------
__global__ void __launch_bounds__(32)
attn_kernel(const bf16_t* __restrict__ qb, const bf16_t* __restrict__ kb,
            const bf16_t* __restrict__ vT, const unsigned char* __restrict__ mask,
            bf16_t* __restrict__ ctx) {
    __shared__ float  sco[16 * S_];       // 32 KB; low half reused for bf16 probs
    __shared__ bf16_t vbuf[HD_ * S_];     // 64 KB staged V^T
    int it = blockIdx.x;                  // 0..31 row tile
    int n  = blockIdx.y;                  // head
    int b  = blockIdx.z;                  // batch
    int lane = threadIdx.x;
    size_t hb = (size_t)b * N_ + n;
    const bf16_t* qm = qb + hb * S_ * HD_;
    const bf16_t* km = kb + hb * S_ * HD_;
    const bf16_t* vm = vT + hb * HD_ * S_;
    int i0 = it * 16;
    int colL = lane & 15;
    int rowAdd = (lane >> 4) << 3;

    // ---- async stage V^T into LDS: 128 x (32 lanes x 16B) = 64 KB ----
    {
        const char* gsrc = (const char*)vm;
        unsigned lbase = (unsigned)(unsigned long long)vbuf;  // LDS offset = addr[31:0]
        for (int i = 0; i < 128; ++i) {
            unsigned off = (unsigned)((i * 32 + lane) * 16);
            unsigned long long ga = (unsigned long long)(gsrc + off);
            unsigned la = lbase + off;
            asm volatile("global_load_async_to_lds_b128 %0, %1, off"
                         :: "v"(la), "v"(ga) : "memory");
        }
    }

    // ---- scores = Q @ K^T / 8, masked -> LDS (ping-pong over K tiles) ----
    v16bf aq0 = load_frag(qm, HD_, i0, 0);
    v16bf aq1 = load_frag(qm, HD_, i0, 32);

    auto store_scores = [&](v8f acc, int jt) {
#pragma unroll
        for (int r = 0; r < 8; ++r) {
            int iL = r + rowAdd;
            int j  = jt * 16 + colL;
            float sv = acc[r] * 0.125f;               // 1/sqrt(64)
            if (!mask[(size_t)(i0 + iL) * S_ + j]) sv = -3.402823466e38f;
            sco[iL * S_ + j] = sv;
        }
    };

    v16bf k0a = load_frag(km, HD_, 0, 0);
    v16bf k1a = load_frag(km, HD_, 0, 32);
    v16bf k0b = load_frag(km, HD_, 16, 0);
    v16bf k1b = load_frag(km, HD_, 16, 32);
    for (int jt = 0; jt < 30; jt += 2) {
        v8f acc = {};
        acc = WMMA_BF16(aq0, k0a, acc);
        acc = WMMA_BF16(aq1, k1a, acc);
        store_scores(acc, jt);
        k0a = load_frag(km, HD_, (jt + 2) * 16, 0);
        k1a = load_frag(km, HD_, (jt + 2) * 16, 32);
        v8f acc2 = {};
        acc2 = WMMA_BF16(aq0, k0b, acc2);
        acc2 = WMMA_BF16(aq1, k1b, acc2);
        store_scores(acc2, jt + 1);
        k0b = load_frag(km, HD_, (jt + 3) * 16, 0);
        k1b = load_frag(km, HD_, (jt + 3) * 16, 32);
    }
    {
        v8f acc = {};
        acc = WMMA_BF16(aq0, k0a, acc);
        acc = WMMA_BF16(aq1, k1a, acc);
        store_scores(acc, 30);
        v8f acc2 = {};
        acc2 = WMMA_BF16(aq0, k0b, acc2);
        acc2 = WMMA_BF16(aq1, k1b, acc2);
        store_scores(acc2, 31);
    }
    __syncthreads();

    // ---- row softmax; write bf16 probs aliased over the score buffer ----
    // prob row rI (1KB at byte rI*1024) only overlaps sco rows <= rI/2, which
    // are fully consumed before the write (LDS ops in-order within a wave).
    bf16_t* prob = reinterpret_cast<bf16_t*>(sco);
    for (int rI = 0; rI < 16; ++rI) {
        float m = -3.402823466e38f;
#pragma unroll
        for (int t = 0; t < 16; ++t) m = fmaxf(m, sco[rI * S_ + lane + 32 * t]);
        for (int off = 16; off > 0; off >>= 1) m = fmaxf(m, __shfl_xor(m, off, 32));
        float e[16];
        float sum = 0.f;
#pragma unroll
        for (int t = 0; t < 16; ++t) {
            e[t] = __expf(sco[rI * S_ + lane + 32 * t] - m);   // masked -> 0
            sum += e[t];
        }
        for (int off = 16; off > 0; off >>= 1) sum += __shfl_xor(sum, off, 32);
        float inv = 1.f / sum;
#pragma unroll
        for (int t = 0; t < 16; ++t)
            prob[rI * S_ + lane + 32 * t] = (bf16_t)(e[t] * inv);
    }
    __syncthreads();

    // ---- wait for the async V^T stage, then ctx = probs @ V from LDS ----
    asm volatile("s_wait_asynccnt 0x0" ::: "memory");

    const bf16_t* probp = reinterpret_cast<const bf16_t*>(sco);
    v8f cacc[4] = {};
    for (int jt = 0; jt < 16; ++jt) {                 // K = 512, step 32
        v16bf a = load_frag(probp, S_, 0, jt * 32);
#pragma unroll
        for (int dt = 0; dt < 4; ++dt) {
            v16bf bv = load_frag(vbuf, S_, dt * 16, jt * 32);  // B col d = row d of V^T
            cacc[dt] = WMMA_BF16(a, bv, cacc[dt]);
        }
    }
#pragma unroll
    for (int dt = 0; dt < 4; ++dt) {
#pragma unroll
        for (int r = 0; r < 8; ++r) {
            int sg = i0 + r + rowAdd;
            int d  = dt * 16 + colL;
            ctx[((size_t)sg * B_ + b) * H_ + n * HD_ + d] = (bf16_t)cacc[dt][r];
        }
    }
}

// ---------------- Kernel 4: output projection (ping-pong pipelined) --------
__global__ void __launch_bounds__(32)
out_gemm_kernel(const bf16_t* __restrict__ ctx, const bf16_t* __restrict__ WoutT,
                const float* __restrict__ bout, float* __restrict__ out) {
    int strip = blockIdx.x;     // 0..15 (64-wide strip)
    int mt    = blockIdx.y;     // 0..511 (16-row tile of S*B)
    int lane  = threadIdx.x;
    int m0 = mt * 16;
    int c0 = strip * 64;

    v8f acc[4] = {};
    v16bf a0 = load_frag(ctx, H_, m0, 0);
    v16bf b0[4];
#pragma unroll
    for (int t = 0; t < 4; ++t) b0[t] = load_frag(WoutT, H_, c0 + 16 * t, 0);
    v16bf a1 = load_frag(ctx, H_, m0, 32);
    v16bf b1[4];
#pragma unroll
    for (int t = 0; t < 4; ++t) b1[t] = load_frag(WoutT, H_, c0 + 16 * t, 32);

    for (int k0 = 0; k0 < H_ - 64; k0 += 64) {
#pragma unroll
        for (int t = 0; t < 4; ++t) acc[t] = WMMA_BF16(a0, b0[t], acc[t]);
        a0 = load_frag(ctx, H_, m0, k0 + 64);
#pragma unroll
        for (int t = 0; t < 4; ++t) b0[t] = load_frag(WoutT, H_, c0 + 16 * t, k0 + 64);
#pragma unroll
        for (int t = 0; t < 4; ++t) acc[t] = WMMA_BF16(a1, b1[t], acc[t]);
        a1 = load_frag(ctx, H_, m0, k0 + 96);
#pragma unroll
        for (int t = 0; t < 4; ++t) b1[t] = load_frag(WoutT, H_, c0 + 16 * t, k0 + 96);
    }
#pragma unroll
    for (int t = 0; t < 4; ++t) acc[t] = WMMA_BF16(a0, b0[t], acc[t]);
#pragma unroll
    for (int t = 0; t < 4; ++t) acc[t] = WMMA_BF16(a1, b1[t], acc[t]);

    int colL = lane & 15;
    int rowAdd = (lane >> 4) << 3;
#pragma unroll
    for (int t = 0; t < 4; ++t) {
#pragma unroll
        for (int r = 0; r < 8; ++r) {
            int mg = m0 + r + rowAdd;
            int c  = c0 + 16 * t + colL;
            out[(size_t)mg * H_ + c] = acc[t][r] + bout[c];
        }
    }
}

// ---------------------------------------------------------------------------
extern "C" void kernel_launch(void* const* d_in, const int* in_sizes, int n_in,
                              void* d_out, int out_size, void* d_ws, size_t ws_size,
                              hipStream_t stream) {
    const float*         hs   = (const float*)d_in[0];          // (S,B,H)
    const unsigned char* mask = (const unsigned char*)d_in[1];  // (S,S) bool
    const float*         Wqkv = (const float*)d_in[2];          // (H,3H)
    const float*         bqkv = (const float*)d_in[3];          // (3H)
    const float*         Wout = (const float*)d_in[4];          // (H,H)
    const float*         bout = (const float*)d_in[5];          // (H)
    float* out = (float*)d_out;                                 // (S,B,H)

    char* ws = (char*)d_ws;
    size_t off = 0;
    bf16_t* xT     = (bf16_t*)(ws + off); off += (size_t)B_ * S_ * H_ * 2;        // 16 MB
    bf16_t* WqkvT  = (bf16_t*)(ws + off); off += (size_t)3 * H_ * H_ * 2;         // 6 MB
    bf16_t* WoutT  = (bf16_t*)(ws + off); off += (size_t)H_ * H_ * 2;             // 2 MB
    bf16_t* qb     = (bf16_t*)(ws + off); off += (size_t)B_ * N_ * S_ * HD_ * 2;  // 16 MB
    bf16_t* kb     = (bf16_t*)(ws + off); off += (size_t)B_ * N_ * S_ * HD_ * 2;  // 16 MB
    bf16_t* vT     = (bf16_t*)(ws + off); off += (size_t)B_ * N_ * HD_ * S_ * 2;  // 16 MB
    bf16_t* ctx    = (bf16_t*)(ws + off); off += (size_t)S_ * B_ * H_ * 2;        // 16 MB
    (void)ws_size; (void)in_sizes; (void)n_in; (void)out_size;

    // 1) precision / layout conversion
    {
        size_t ne = (size_t)S_ * B_ * H_;
        conv_x_kernel<<<dim3((unsigned)((ne + 255) / 256)), 256, 0, stream>>>(hs, xT);
    }
    {
        size_t ne = (size_t)H_ * 3 * H_;
        conv_wT_kernel<<<dim3((unsigned)((ne + 255) / 256)), 256, 0, stream>>>(Wqkv, WqkvT, H_, 3 * H_);
    }
    {
        size_t ne = (size_t)H_ * H_;
        conv_wT_kernel<<<dim3((unsigned)((ne + 255) / 256)), 256, 0, stream>>>(Wout, WoutT, H_, H_);
    }

    // 2) QKV projection
    qkv_gemm_kernel<<<dim3(48, 32, 16), 32, 0, stream>>>(xT, WqkvT, bqkv, qb, kb, vT);

    // 3) masked attention per (row-tile, head, batch)
    attn_kernel<<<dim3(32, 16, 16), 32, 0, stream>>>(qb, kb, vT, mask, ctx);

    // 4) output projection
    out_gemm_kernel<<<dim3(16, 512), 32, 0, stream>>>(ctx, WoutT, bout, out);
}
